// SanaTransformerBlock_90099823935980
// MI455X (gfx1250) — compile-verified, matrix operand order checked by reading, not run on hardware
//
#include <hip/hip_runtime.h>

typedef __bf16 bf16;
typedef __attribute__((ext_vector_type(8)))  __bf16 v8bf;
typedef __attribute__((ext_vector_type(16))) __bf16 v16bf;
typedef __attribute__((ext_vector_type(8)))  float  v8f;

#define B_   2
#define S_   4096
#define D_   2240
#define H1_  70
#define D1_  32
#define H2_  20
#define D2_  112
#define L_   300
#define HID_ 5600

// ---------------- small helpers ----------------
__device__ __forceinline__ v8f v8f_zero() {
  v8f x;
#pragma unroll
  for (int i = 0; i < 8; ++i) x[i] = 0.f;
  return x;
}
__device__ __forceinline__ v8bf v8bf_zero() {
  v8bf x;
#pragma unroll
  for (int i = 0; i < 8; ++i) x[i] = (bf16)0.f;
  return x;
}

__device__ __forceinline__ float siluf(float v) { return v / (1.f + __expf(-v)); }

// 16-bit WMMA fragment: lanes 0-15 hold K={0..7,16..23}, lanes 16-31 K={8..15,24..31}
// base points at the start of a 32-element K-window for this lane's row/col in LDS.
__device__ __forceinline__ v16bf ldfrag(const bf16* base, int lane) {
  const int k0 = (lane >> 4) << 3;
  v8bf lo = *(const v8bf*)(base + k0);
  v8bf hi = *(const v8bf*)(base + k0 + 16);
  return __builtin_shufflevector(lo, hi, 0,1,2,3,4,5,6,7,8,9,10,11,12,13,14,15);
}

__device__ __forceinline__ v8f wmma_bf16(v16bf a, v16bf b, v8f c) {
  return __builtin_amdgcn_wmma_f32_16x16x32_bf16(false, a, false, b, (short)0, c,
                                                 false, false);
}

// Flat->LDS 32-bit offset (LDS flat aperture keeps the byte offset in addr[31:0]).
__device__ __forceinline__ unsigned ldsOff(const void* p) {
  return (unsigned)(unsigned long long)p;
}
// CDNA5 async global->LDS 16-byte copy (tracked by ASYNCcnt).
__device__ __forceinline__ void g2l_b128(void* lds, const void* g) {
  asm volatile("global_load_async_to_lds_b128 %0, %1, off"
               :: "v"(ldsOff(lds)), "v"(g) : "memory");
}
__device__ __forceinline__ void wait_async0() {
  asm volatile("s_wait_asynccnt 0" ::: "memory");
}

// ---------------- conversion kernels ----------------
__global__ __launch_bounds__(256) void k_cvt(const float* __restrict__ src,
                                             bf16* __restrict__ dst, long long n) {
  long long i = (long long)blockIdx.x * 256 + threadIdx.x;
  long long stride = (long long)gridDim.x * 256;
  for (; i < n; i += stride) dst[i] = (bf16)src[i];
}

// W[K][N] fp32 -> Wt[N][K] bf16 (so GEMM B-tiles stage contiguously per column)
__global__ __launch_bounds__(256) void k_cvt_t(const float* __restrict__ W,
                                               bf16* __restrict__ Wt, int K, int N) {
  __shared__ float tile[32][33];
  int kb = blockIdx.y * 32, nb = blockIdx.x * 32;
  int tx = threadIdx.x & 31, ty = threadIdx.x >> 5;
  for (int i = ty; i < 32; i += 8) {
    int k = kb + i, n = nb + tx;
    tile[i][tx] = (k < K && n < N) ? W[(long long)k * N + n] : 0.f;
  }
  __syncthreads();
  for (int i = ty; i < 32; i += 8) {
    int n = nb + i, k = kb + tx;
    if (n < N && k < K) Wt[(long long)n * K + k] = (bf16)tile[tx][i];
  }
}

// mods[b,6,d] = sst[6,d] + timestep[b, 6*d]
__global__ __launch_bounds__(256) void k_mods(const float* __restrict__ sst,
                                              const float* __restrict__ ts,
                                              float* __restrict__ mods) {
  long long i = (long long)blockIdx.x * 256 + threadIdx.x;
  if (i < (long long)B_ * 6 * D_) mods[i] = sst[i % (6LL * D_)] + ts[i];
}

// LayerNorm + AdaLN affine -> bf16
__global__ __launch_bounds__(256) void k_ln_mod(const float* __restrict__ x,
                                                const float* __restrict__ mods,
                                                int scaleIdx, int shiftIdx,
                                                bf16* __restrict__ out) {
  int row = blockIdx.x;                 // 0..B*S-1
  int b = row / S_;
  const float* xr = x + (long long)row * D_;
  float s = 0.f, s2 = 0.f;
  for (int i = threadIdx.x; i < D_; i += 256) { float v = xr[i]; s += v; s2 += v * v; }
  __shared__ float red0[8], red1[8];
  for (int o = 16; o > 0; o >>= 1) { s += __shfl_down(s, o, 32); s2 += __shfl_down(s2, o, 32); }
  int wid = threadIdx.x >> 5, lane = threadIdx.x & 31;
  if (lane == 0) { red0[wid] = s; red1[wid] = s2; }
  __syncthreads();
  if (threadIdx.x == 0) {
    float a = 0.f, c = 0.f;
    for (int i = 0; i < 8; ++i) { a += red0[i]; c += red1[i]; }
    float m = a / D_;
    red0[0] = m; red1[0] = c / D_ - m * m;
  }
  __syncthreads();
  float mean = red0[0];
  float rstd = rsqrtf(red1[0] + 1e-6f);
  const float* sc = mods + (long long)b * 6 * D_ + (long long)scaleIdx * D_;
  const float* sh = mods + (long long)b * 6 * D_ + (long long)shiftIdx * D_;
  bf16* o = out + (long long)row * D_;
  for (int i = threadIdx.x; i < D_; i += 256)
    o[i] = (bf16)(((xr[i] - mean) * rstd) * (1.f + sc[i]) + sh[i]);
}

// ---------------- WMMA GEMM: C = act(A[M,K] @ W[K,N] + bias) (+residual/gate) ----
// A bf16 row-major, Wt bf16 [N][K]; block tile 128x64, 8 waves of 32x32, K-step 32.
__global__ __launch_bounds__(256, 1) void k_gemm(
    const bf16* __restrict__ A, const bf16* __restrict__ Wt,
    const float* __restrict__ bias, const float* __restrict__ addsrc,
    const float* __restrict__ gmods, int gateIdx, int rowsPerBatch,
    float* __restrict__ Cf, bf16* __restrict__ Cb,
    int M, int N, int K, int act) {
  __shared__ bf16 As[2][128][32];     // 16 KB
  __shared__ bf16 Bs[2][64][32];      //  8 KB
  const int t = threadIdx.x;
  const int lane = t & 31;
  const int w = t >> 5;               // 0..7
  const int wm = w & 3, wn = w >> 2;  // 4x2 wave grid
  const long long mBase = (long long)blockIdx.y * 128;
  const long long nBase = (long long)blockIdx.x * 64;
  const int KT = K / 32;

  auto stage = [&](int buf, int kt) {
    const long long kOff = (long long)kt * 32;
    // A: 128 rows x 64B -> 512 x 16B chunks; 2 per thread
#pragma unroll
    for (int c = t; c < 512; c += 256) {
      int r = c >> 2, j = c & 3;
      long long row = mBase + r; if (row >= M) row = M - 1;
      g2l_b128(&As[buf][r][j * 8], A + row * (long long)K + kOff + j * 8);
    }
    // B: 64 rows x 64B -> 256 x 16B chunks; 1 per thread
    {
      int r = t >> 2, j = t & 3;
      g2l_b128(&Bs[buf][r][j * 8], Wt + (nBase + r) * (long long)K + kOff + j * 8);
    }
    if (kt + 2 < KT) {  // L2 prefetch of the tile after next
      long long row = mBase + (t >> 2); if (row >= M) row = M - 1;
      __builtin_prefetch(A + row * (long long)K + kOff + 64, 0, 1);
    }
  };

  stage(0, 0);
  wait_async0();
  __syncthreads();

  v8f acc00 = v8f_zero(), acc01 = v8f_zero(), acc10 = v8f_zero(), acc11 = v8f_zero();
  for (int kt = 0; kt < KT; ++kt) {
    const int cur = kt & 1;
    if (kt + 1 < KT) stage(cur ^ 1, kt + 1);

    v16bf a0 = ldfrag(&As[cur][wm * 32 + (lane & 15)][0], lane);
    v16bf a1 = ldfrag(&As[cur][wm * 32 + 16 + (lane & 15)][0], lane);
    v16bf b0 = ldfrag(&Bs[cur][wn * 32 + (lane & 15)][0], lane);
    v16bf b1 = ldfrag(&Bs[cur][wn * 32 + 16 + (lane & 15)][0], lane);
    acc00 = wmma_bf16(a0, b0, acc00);
    acc01 = wmma_bf16(a0, b1, acc01);
    acc10 = wmma_bf16(a1, b0, acc10);
    acc11 = wmma_bf16(a1, b1, acc11);

    wait_async0();
    __syncthreads();
  }

  // epilogue (C layout: col = lane&15, rows = r + 8*(lane>>4) within each 16x16)
  v8f accs[2][2] = {{acc00, acc01}, {acc10, acc11}};
#pragma unroll
  for (int mt = 0; mt < 2; ++mt)
#pragma unroll
    for (int nt = 0; nt < 2; ++nt) {
      int col = (int)nBase + wn * 32 + nt * 16 + (lane & 15);
      long long rowB = mBase + wm * 32 + mt * 16 + ((lane >> 4) << 3);
      float bv = bias ? bias[col] : 0.f;
#pragma unroll
      for (int r = 0; r < 8; ++r) {
        long long row = rowB + r;
        if (row < M) {
          float v = accs[mt][nt][r] + bv;
          if (act == 1) v = fmaxf(v, 0.f);
          else if (act == 2) v = siluf(v);
          if (gmods && gateIdx >= 0) {
            long long b = row / rowsPerBatch;
            float g = gmods[b * 6LL * N + (long long)gateIdx * N + col];
            v = addsrc[row * (long long)N + col] + g * v;
          } else if (addsrc) {
            v = addsrc[row * (long long)N + col] + v;
          }
          if (Cf) Cf[row * (long long)N + col] = v;
          if (Cb) Cb[row * (long long)N + col] = (bf16)v;
        }
      }
    }
}

// ---------------- linear attention: scores[b,h,33,32] = sum_s vpad[s,e]*k[s,d] ----
__global__ __launch_bounds__(256) void k_scores(const bf16* __restrict__ kB,
                                                const bf16* __restrict__ vB,
                                                float* __restrict__ scores) {
  int bh = blockIdx.x;                       // B*H1
  int b = bh / H1_, h = bh % H1_;
  __shared__ float ks[32][33];
  __shared__ float vs[32][33];
  int t = threadIdx.x;
  int e = t >> 3;                            // 0..31
  int d0 = (t & 7) * 4;
  float acc[4] = {0.f, 0.f, 0.f, 0.f};
  float acc1[4] = {0.f, 0.f, 0.f, 0.f};      // "ones" row (e==0 threads track)
  for (int s0 = 0; s0 < S_; s0 += 32) {
    for (int i = t; i < 1024; i += 256) {
      int s = i >> 5, d = i & 31;
      long long base = ((long long)b * S_ + s0 + s) * D_ + h * 32 + d;
      ks[s][d] = (float)kB[base];
      vs[s][d] = (float)vB[base];
    }
    __syncthreads();
    for (int s = 0; s < 32; ++s) {
      float ve = vs[s][e];
#pragma unroll
      for (int j = 0; j < 4; ++j) acc[j] += ve * ks[s][d0 + j];
      if (e == 0) {
#pragma unroll
        for (int j = 0; j < 4; ++j) acc1[j] += ks[s][d0 + j];
      }
    }
    __syncthreads();
  }
  long long base = (long long)bh * 33 * 32;
#pragma unroll
  for (int j = 0; j < 4; ++j) scores[base + e * 32 + d0 + j] = acc[j];
  if (e == 0) {
#pragma unroll
    for (int j = 0; j < 4; ++j) scores[base + 32 * 32 + d0 + j] = acc1[j];
  }
}

// attn[b,s,h*32+e] = (q . scores[e,:]) / (q . scores[32,:] + 1e-15)
__global__ __launch_bounds__(256) void k_apply(const bf16* __restrict__ qB,
                                               const float* __restrict__ scores,
                                               bf16* __restrict__ attnB) {
  int bh = blockIdx.x;
  int b = bh / H1_, h = bh % H1_;
  int s = blockIdx.y * 256 + threadIdx.x;
  __shared__ float sc[33][32];
  for (int i = threadIdx.x; i < 33 * 32; i += 256)
    sc[i >> 5][i & 31] = scores[(long long)bh * 33 * 32 + i];
  __syncthreads();
  float q[32];
  const bf16* qp = qB + ((long long)b * S_ + s) * D_ + h * 32;
#pragma unroll
  for (int d = 0; d < 32; ++d) q[d] = (float)qp[d];
  float den = 1e-15f;
#pragma unroll
  for (int d = 0; d < 32; ++d) den += sc[32][d] * q[d];
  float inv = 1.f / den;
  bf16* op = attnB + ((long long)b * S_ + s) * D_ + h * 32;
  for (int e = 0; e < 32; ++e) {
    float a = 0.f;
#pragma unroll
    for (int d = 0; d < 32; ++d) a += sc[e][d] * q[d];
    op[e] = (bf16)(a * inv);
  }
}

// ---------------- flash cross-attention (WMMA QK^T and P.V, online softmax) ----
__global__ __launch_bounds__(256, 1) void k_xattn(const bf16* __restrict__ q2B,
                                                  const bf16* __restrict__ k2B,
                                                  const bf16* __restrict__ v2B,
                                                  bf16* __restrict__ attn2B) {
  const int bh = blockIdx.x;           // B*H2
  const int b = bh / H2_, h = bh % H2_;
  const int s0 = blockIdx.y * 32;
  __shared__ bf16 Qs[32][128];         // 8 KB (D2=112 zero-padded to 128)
  __shared__ bf16 Ks[64][128];         // 16 KB context chunk
  __shared__ bf16 VT[112][64];         // 14 KB V^T chunk
  __shared__ float Ls[32][64];         // 8 KB logits
  __shared__ bf16 Ps[32][64];          // 4 KB probs
  __shared__ float mrow[32], lrow[32], arow[32];
  const int t = threadIdx.x, lane = t & 31, w = t >> 5;
  const int mt = w & 1, nt0 = w >> 1;  // 2 m-tiles; nt0 in 0..3 (+4 for second tile)

  for (int idx = t; idx < 512; idx += 256) {   // Q: 32 x 128
    int r = idx >> 4, c8 = (idx & 15) * 8;
    v8bf val = v8bf_zero();
    if (c8 < D2_)
      val = *(const v8bf*)(q2B + ((long long)b * S_ + s0 + r) * D_ + h * D2_ + c8);
    *(v8bf*)&Qs[r][c8] = val;
  }
  if (t < 32) { mrow[t] = -3e38f; lrow[t] = 0.f; }
  __syncthreads();

  v8f acc0 = v8f_zero(), acc1 = v8f_zero();
  for (int c0 = 0; c0 < 320; c0 += 64) {
    for (int idx = t; idx < 64 * 16; idx += 256) {   // K chunk: 64 x 128
      int r = idx >> 4, c8 = (idx & 15) * 8;
      int l = c0 + r;
      v8bf val = v8bf_zero();
      if (l < L_ && c8 < D2_)
        val = *(const v8bf*)(k2B + ((long long)b * L_ + l) * D_ + h * D2_ + c8);
      *(v8bf*)&Ks[r][c8] = val;
    }
    for (int idx = t; idx < 112 * 64; idx += 256) {  // V^T chunk: 112 x 64
      int d = idx >> 6, i = idx & 63;
      int l = c0 + i;
      VT[d][i] = (l < L_) ? v2B[((long long)b * L_ + l) * D_ + h * D2_ + d] : (bf16)0.f;
    }
    __syncthreads();
    {                                               // QK^T: one 16x16 tile per wave
      int ntq = w >> 1;                             // 0..3
      v8f lg = v8f_zero();
#pragma unroll
      for (int kk = 0; kk < 4; ++kk) {
        v16bf a = ldfrag(&Qs[mt * 16 + (lane & 15)][kk * 32], lane);
        v16bf bb = ldfrag(&Ks[ntq * 16 + (lane & 15)][kk * 32], lane);
        lg = wmma_bf16(a, bb, lg);
      }
      const float scale = 0.0944911183f;            // 112^-0.5
      int col = ntq * 16 + (lane & 15);
      int rbase = mt * 16 + ((lane >> 4) << 3);
#pragma unroll
      for (int r = 0; r < 8; ++r) Ls[rbase + r][col] = lg[r] * scale;
    }
    __syncthreads();
    if (t < 32) {                                   // online softmax per row
      float m = mrow[t];
      int lim = L_ - c0; if (lim > 64) lim = 64;
      float cm = m;
      for (int j = 0; j < lim; ++j) cm = fmaxf(cm, Ls[t][j]);
      float alpha = __expf(m - cm), sum = 0.f;
      for (int j = 0; j < 64; ++j) {
        float p = (j < lim) ? __expf(Ls[t][j] - cm) : 0.f;
        Ps[t][j] = (bf16)p; sum += p;
      }
      mrow[t] = cm; lrow[t] = lrow[t] * alpha + sum; arow[t] = alpha;
    }
    __syncthreads();
    {                                               // rescale + P.V
      int rbase = mt * 16 + ((lane >> 4) << 3);
#pragma unroll
      for (int r = 0; r < 8; ++r) { float al = arow[rbase + r]; acc0[r] *= al; acc1[r] *= al; }
#pragma unroll
      for (int kk = 0; kk < 2; ++kk) {
        v16bf a = ldfrag(&Ps[mt * 16 + (lane & 15)][kk * 32], lane);
        v16bf b0 = ldfrag(&VT[nt0 * 16 + (lane & 15)][kk * 32], lane);
        acc0 = wmma_bf16(a, b0, acc0);
        if (nt0 < 3) {
          v16bf b1 = ldfrag(&VT[(nt0 + 4) * 16 + (lane & 15)][kk * 32], lane);
          acc1 = wmma_bf16(a, b1, acc1);
        }
      }
    }
    __syncthreads();
  }
  int rbase = mt * 16 + ((lane >> 4) << 3);
#pragma unroll
  for (int r = 0; r < 8; ++r) {
    float inv = 1.f / (lrow[rbase + r] + 1e-20f);
    long long srow = s0 + rbase + r;
    long long base = ((long long)b * S_ + srow) * D_ + h * D2_;
    attn2B[base + nt0 * 16 + (lane & 15)] = (bf16)(acc0[r] * inv);
    if (nt0 < 3)
      attn2B[base + (nt0 + 4) * 16 + (lane & 15)] = (bf16)(acc1[r] * inv);
  }
}

// ---------------- depthwise 3x3 + bias + GLU (NHWC, channels fastest) ----------
__global__ __launch_bounds__(256) void k_dwglu(const bf16* __restrict__ ff1,
                                               const float* __restrict__ cdw,
                                               const float* __restrict__ cdb,
                                               bf16* __restrict__ glu) {
  long long idx = (long long)blockIdx.x * 256 + threadIdx.x;
  const long long total = (long long)B_ * 64 * 64 * HID_;
  if (idx >= total) return;
  int c = (int)(idx % HID_);
  long long rest = idx / HID_;
  int x = (int)(rest % 64); rest /= 64;
  int y = (int)(rest % 64);
  int b = (int)(rest / 64);
  float a = cdb[c], g = cdb[c + HID_];
#pragma unroll
  for (int dy = -1; dy <= 1; ++dy)
#pragma unroll
    for (int dx = -1; dx <= 1; ++dx) {
      int yy = y + dy, xx = x + dx;
      if (yy >= 0 && yy < 64 && xx >= 0 && xx < 64) {
        long long base = (((long long)b * 64 + yy) * 64 + xx) * (2LL * HID_);
        int wi = (dy + 1) * 3 + (dx + 1);
        a += cdw[(long long)c * 9 + wi] * (float)ff1[base + c];
        g += cdw[((long long)c + HID_) * 9 + wi] * (float)ff1[base + c + HID_];
      }
    }
  glu[idx] = (bf16)(a * siluf(g));
}

// ---------------- host orchestration ----------------
extern "C" void kernel_launch(void* const* d_in, const int* in_sizes, int n_in,
                              void* d_out, int out_size, void* d_ws, size_t ws_size,
                              hipStream_t stream) {
  (void)in_sizes; (void)n_in; (void)out_size; (void)ws_size;
  const float* hidden = (const float*)d_in[0];
  const float* enc    = (const float*)d_in[2];
  const float* ts     = (const float*)d_in[4];
  const float* sst    = (const float*)d_in[7];
  const float* wsrc[8] = {(const float*)d_in[8],  (const float*)d_in[10],
                          (const float*)d_in[12], (const float*)d_in[14],
                          (const float*)d_in[16], (const float*)d_in[18],
                          (const float*)d_in[20], (const float*)d_in[22]};
  const float* bq1 = (const float*)d_in[9];  const float* bk1 = (const float*)d_in[11];
  const float* bv1 = (const float*)d_in[13]; const float* bo1 = (const float*)d_in[15];
  const float* bq2 = (const float*)d_in[17]; const float* bk2 = (const float*)d_in[19];
  const float* bv2 = (const float*)d_in[21]; const float* bo2 = (const float*)d_in[23];
  const float* ciw = (const float*)d_in[24]; const float* cib = (const float*)d_in[25];
  const float* cdw = (const float*)d_in[26]; const float* cdb = (const float*)d_in[27];
  const float* cpw = (const float*)d_in[28];

  char* ws = (char*)d_ws; size_t off = 0;
  auto alloc = [&](size_t bytes) -> void* {
    void* p = ws + off; off += (bytes + 255) & ~(size_t)255; return p;
  };
  const long long BS = (long long)B_ * S_;
  const size_t ND2 = (size_t)BS * D_ * 2;      // bf16 activation
  const size_t ND4 = (size_t)BS * D_ * 4;      // fp32 activation

  bf16* wT[8];
  for (int i = 0; i < 8; ++i) wT[i] = (bf16*)alloc((size_t)D_ * D_ * 2);
  bf16* ciW   = (bf16*)alloc((size_t)2 * HID_ * D_ * 2);
  bf16* cpW   = (bf16*)alloc((size_t)HID_ * D_ * 2);
  bf16* encB  = (bf16*)alloc((size_t)B_ * L_ * D_ * 2);
  float* mods = (float*)alloc((size_t)B_ * 6 * D_ * 4);
  bf16* hB    = (bf16*)alloc(ND2);
  bf16* qB    = (bf16*)alloc(ND2);
  bf16* kB    = (bf16*)alloc(ND2);
  bf16* vB    = (bf16*)alloc(ND2);
  float* scor = (float*)alloc((size_t)B_ * H1_ * 33 * 32 * 4);
  bf16* attnB = (bf16*)alloc(ND2);
  float* xf   = (float*)alloc(ND4);
  bf16* xb    = (bf16*)alloc(ND2);
  bf16* q2B   = (bf16*)alloc(ND2);
  bf16* k2B   = (bf16*)alloc((size_t)B_ * L_ * D_ * 2);
  bf16* v2B   = (bf16*)alloc((size_t)B_ * L_ * D_ * 2);
  bf16* at2B  = (bf16*)alloc(ND2);
  float* x2f  = (float*)alloc(ND4);
  bf16* ff1B  = (bf16*)alloc((size_t)BS * 2 * HID_ * 2);
  bf16* gluB  = (bf16*)alloc((size_t)BS * HID_ * 2);

  dim3 blk(256);
  for (int i = 0; i < 8; ++i)
    k_cvt_t<<<dim3(D_ / 32, D_ / 32), blk, 0, stream>>>(wsrc[i], wT[i], D_, D_);
  k_cvt<<<4096, blk, 0, stream>>>(ciw, ciW, (long long)2 * HID_ * D_);
  k_cvt<<<4096, blk, 0, stream>>>(cpw, cpW, (long long)HID_ * D_);
  k_cvt<<<1024, blk, 0, stream>>>(enc, encB, (long long)B_ * L_ * D_);
  k_mods<<<(B_ * 6 * D_ + 255) / 256, blk, 0, stream>>>(sst, ts, mods);
  k_ln_mod<<<(int)BS, blk, 0, stream>>>(hidden, mods, 1, 0, hB);

  auto gemm = [&](const bf16* A, const bf16* W, const float* bias, const float* add,
                  const float* gm, int gi, float* Cf, bf16* Cb,
                  int M, int N, int K, int act) {
    k_gemm<<<dim3(N / 64, (M + 127) / 128), blk, 0, stream>>>(
        A, W, bias, add, gm, gi, S_, Cf, Cb, M, N, K, act);
  };

  // self-attention projections (ReLU-kernelized q/k)
  gemm(hB, wT[0], bq1, nullptr, nullptr, -1, nullptr, qB, (int)BS, D_, D_, 1);
  gemm(hB, wT[1], bk1, nullptr, nullptr, -1, nullptr, kB, (int)BS, D_, D_, 1);
  gemm(hB, wT[2], bv1, nullptr, nullptr, -1, nullptr, vB, (int)BS, D_, D_, 0);
  k_scores<<<B_ * H1_, blk, 0, stream>>>(kB, vB, scor);
  k_apply<<<dim3(B_ * H1_, S_ / 256), blk, 0, stream>>>(qB, scor, attnB);
  // x = hidden + gate_msa * (attn @ wo1 + bo1)
  gemm(attnB, wT[3], bo1, hidden, mods, 2, xf, xb, (int)BS, D_, D_, 0);
  // cross attention
  gemm(xb, wT[4], bq2, nullptr, nullptr, -1, nullptr, q2B, (int)BS, D_, D_, 0);
  gemm(encB, wT[5], bk2, nullptr, nullptr, -1, nullptr, k2B, B_ * L_, D_, D_, 0);
  gemm(encB, wT[6], bv2, nullptr, nullptr, -1, nullptr, v2B, B_ * L_, D_, D_, 0);
  k_xattn<<<dim3(B_ * H2_, S_ / 32), blk, 0, stream>>>(q2B, k2B, v2B, at2B);
  // x = (attn2 @ wo2 + bo2) + x
  gemm(at2B, wT[7], bo2, xf, nullptr, -1, x2f, nullptr, (int)BS, D_, D_, 0);
  // GLUMBConv FFN
  k_ln_mod<<<(int)BS, blk, 0, stream>>>(x2f, mods, 4, 3, hB);
  gemm(hB, ciW, cib, nullptr, nullptr, -1, nullptr, ff1B, (int)BS, 2 * HID_, D_, 2);
  long long dwTotal = (long long)B_ * 64 * 64 * HID_;
  k_dwglu<<<(unsigned)((dwTotal + 255) / 256), blk, 0, stream>>>(ff1B, cdw, cdb, gluB);
  // out = x + gate_mlp * (glu @ cp_w)
  gemm(gluB, cpW, nullptr, x2f, mods, 5, (float*)d_out, nullptr, (int)BS, D_, HID_, 0);
}
